// DeepDFTinv_Message_WithReciever_2619930050971
// MI455X (gfx1250) — matrix operand order repaired and checked by reading, not compiled, not run on hardware
//
#include <hip/hip_runtime.h>

typedef __attribute__((ext_vector_type(16))) _Float16 v16h;
typedef __attribute__((ext_vector_type(8)))  _Float16 v8h;
typedef __attribute__((ext_vector_type(8)))  float    v8f;

constexpr int NB = 2, NP = 4096, NA = 96, NF = 32, NE = 20;
constexpr int PT = NP / 16;            // 256 p-tiles per batch
constexpr int WAVES_PER_BLOCK = 8;
constexpr int ASPLIT = 4;              // A-axis split for occupancy (2048 waves)
constexpr int NAS = NA / ASPLIT;       // 24 a-values per wave
constexpr int OUT_ELEMS = NB * NP * NF;

__device__ __forceinline__ float ssp_f(float x) {
  // softplus(x) - ln2 via raw hw exp2/log2: log2 arg is in [1,2] -> no fixups needed
  const float t = __builtin_amdgcn_exp2f(-1.442695040888963f * fabsf(x));
  const float l = __builtin_amdgcn_logf(1.0f + t);           // log2(1+t)
  return fmaf(0.6931471805599453f, l, fmaxf(x, 0.0f) - 0.6931471805599453f);
}

// src = scalar @ W1n[:F], tgt = scalar_reciever @ W1n[F:]
__global__ void precompute_proj(const float* __restrict__ scalar,
                                const float* __restrict__ scalar_reciever,
                                const float* __restrict__ W1n,
                                float* __restrict__ tgt, float* __restrict__ src) {
  int idx = blockIdx.x * blockDim.x + threadIdx.x;
  const int nt = NB * NP * NF;
  const int ns = NB * NA * NF;
  if (idx < nt) {
    int f = idx & (NF - 1);
    int row = idx >> 5;
    const float* in = scalar_reciever + (size_t)row * NF;
    float acc = 0.f;
    #pragma unroll
    for (int k = 0; k < NF; ++k) acc += in[k] * W1n[(NF + k) * NF + f];
    tgt[idx] = acc;
  } else if (idx < nt + ns) {
    int j = idx - nt;
    int f = j & (NF - 1);
    int row = j >> 5;
    const float* in = scalar + (size_t)row * NF;
    float acc = 0.f;
    #pragma unroll
    for (int k = 0; k < NF; ++k) acc += in[k] * W1n[k * NF + f];
    src[j] = acc;
  }
}

__global__ __launch_bounds__(256) void deepdft_fused(
    const float* __restrict__ expansion,
    const float* __restrict__ mask,
    const float* __restrict__ edge_distance,
    const float* __restrict__ W1e, const float* __restrict__ b1e,
    const float* __restrict__ W2e, const float* __restrict__ b2e,
    const float* __restrict__ b1n,
    const float* __restrict__ W2n, const float* __restrict__ b2n,
    const float* __restrict__ tgt, const float* __restrict__ src,
    float* __restrict__ part)
{
  const int lane = threadIdx.x & 31;
  const int wv   = threadIdx.x >> 5;
  const int half = lane >> 4;
  const int ln   = lane & 15;

  const int wid   = blockIdx.x * WAVES_PER_BLOCK + wv;   // 0..2047
  const int slice = wid & (ASPLIT - 1);
  const int tile  = wid >> 2;                            // 0..511
  const int b  = tile / PT;
  const int p0 = (tile % PT) * 16;
  const int a0 = slice * NAS;

  __shared__ alignas(16) _Float16 lds_g[WAVES_PER_BLOCK][16][32];
  __shared__ alignas(16) float    lds_w[WAVES_PER_BLOCK][NAS][16];

  // ---- B-matrix (weight) fragments, f16, built once per wave ----
  // B layout: lane holds column N = ln (+ N-tile base); element i -> K = (i&7) + 16*(i>>3) + 8*half.
  v16h Bw1e0, Bw1e1, Bw2e0, Bw2e1, Bw2n0, Bw2n1;
  #pragma unroll
  for (int i = 0; i < 16; ++i) {
    const int k = (i & 7) + 16 * (i >> 3) + 8 * half;
    Bw1e0[i] = (_Float16)((k < NE) ? W1e[k * NF + ln     ] : 0.0f);
    Bw1e1[i] = (_Float16)((k < NE) ? W1e[k * NF + ln + 16] : 0.0f);
    Bw2e0[i] = (_Float16)W2e[k * NF + ln     ];
    Bw2e1[i] = (_Float16)W2e[k * NF + ln + 16];
    Bw2n0[i] = (_Float16)W2n[k * NF + ln     ];
    Bw2n1[i] = (_Float16)W2n[k * NF + ln + 16];
  }

  // ---- hoisted mask * distance-gate for this wave's (p-tile, a-slice) ----
  for (int j = lane; j < NAS * 16; j += 32) {
    const int ar = j >> 4, pl = j & 15;
    const size_t eidx = (size_t)(b * NP + p0 + pl) * NA + (a0 + ar);
    const float m = mask[eidx];
    const float d = edge_distance[eidx];
    // 1 - sigmoid(5*(d-3.5)) == 1/(1+exp2(5*log2e*(d-3.5)))
    lds_w[wv][ar][pl] =
        m * __builtin_amdgcn_rcpf(1.0f + __builtin_amdgcn_exp2f(7.213475204444817f * (d - 3.5f)));
  }

  // ---- per-lane hoisted values ----
  // A-fragment K pattern: element i -> K = (i&7) + 16*(i>>3) + 8*half
  const int p_lane = p0 + ln;
  const int k0 = 8 * half;
  float th[16];                               // tgt[p_lane][k] + b1n[k] (loop invariant)
  {
    const float* tr = tgt + ((size_t)b * NP + p_lane) * NF;
    const float4 t0 = *(const float4*)(tr + k0);
    const float4 t1 = *(const float4*)(tr + k0 + 4);
    const float4 t2 = *(const float4*)(tr + k0 + 16);
    const float4 t3 = *(const float4*)(tr + k0 + 20);
    const float4 u0 = *(const float4*)(b1n + k0);
    const float4 u1 = *(const float4*)(b1n + k0 + 4);
    const float4 u2 = *(const float4*)(b1n + k0 + 16);
    const float4 u3 = *(const float4*)(b1n + k0 + 20);
    th[0]=t0.x+u0.x; th[1]=t0.y+u0.y; th[2]=t0.z+u0.z; th[3]=t0.w+u0.w;
    th[4]=t1.x+u1.x; th[5]=t1.y+u1.y; th[6]=t1.z+u1.z; th[7]=t1.w+u1.w;
    th[8]=t2.x+u2.x; th[9]=t2.y+u2.y; th[10]=t2.z+u2.z; th[11]=t2.w+u2.w;
    th[12]=t3.x+u3.x; th[13]=t3.y+u3.y; th[14]=t3.z+u3.z; th[15]=t3.w+u3.w;
  }
  const float b1e_c0 = b1e[ln], b1e_c1 = b1e[ln + 16];
  const float b2e_c0 = b2e[ln], b2e_c1 = b2e[ln + 16];
  const float b2n_c0 = b2n[ln], b2n_c1 = b2n[ln + 16];

  const float* __restrict__ exp_row0 = expansion + ((size_t)(b * NP + p_lane) * NA) * NE;

  v8f acc0 = {}, acc1 = {};

  #pragma unroll 1
  for (int a = a0; a < a0 + NAS; ++a) {
    // ---- A) expansion fragment (pure 16B loads, branchless tail zeroing) ----
    const float* __restrict__ er = exp_row0 + (size_t)a * NE;
    __builtin_prefetch(er + NE, 0, 1);
    const float4 e0 = *(const float4*)(er + k0);            // K = k0..k0+3
    const float4 e1 = *(const float4*)(er + k0 + 4);        // K = k0+4..k0+7
    const float4 e2 = *(const float4*)(er + 16 - 2 * k0);   // half0: K=16..19; half1: dummy
    const bool hv = (half == 0);
    v16h aE;
    aE[0]=(_Float16)e0.x;  aE[1]=(_Float16)e0.y;  aE[2]=(_Float16)e0.z;  aE[3]=(_Float16)e0.w;
    aE[4]=(_Float16)e1.x;  aE[5]=(_Float16)e1.y;  aE[6]=(_Float16)e1.z;  aE[7]=(_Float16)e1.w;
    aE[8]=(_Float16)(hv ? e2.x : 0.f);  aE[9]=(_Float16)(hv ? e2.y : 0.f);
    aE[10]=(_Float16)(hv ? e2.z : 0.f); aE[11]=(_Float16)(hv ? e2.w : 0.f);
    aE[12]=(_Float16)0.f;  aE[13]=(_Float16)0.f;  aE[14]=(_Float16)0.f;  aE[15]=(_Float16)0.f;

    v8f c10 = {}, c11 = {};
    c10 = __builtin_amdgcn_wmma_f32_16x16x32_f16(false, aE, false, Bw1e0, (short)0, c10, false, false);
    c11 = __builtin_amdgcn_wmma_f32_16x16x32_f16(false, aE, false, Bw1e1, (short)0, c11, false, false);

    // ---- B) independent of GEMM1: node fragment + GEMM3 (fills GEMM1 hazard shadow) ----
    const float* __restrict__ sr = src + (size_t)(b * NA + a) * NF;
    const float4 s0 = *(const float4*)(sr + k0);
    const float4 s1 = *(const float4*)(sr + k0 + 4);
    const float4 s2 = *(const float4*)(sr + k0 + 16);
    const float4 s3 = *(const float4*)(sr + k0 + 20);
    const float hsrc[16] = {s0.x, s0.y, s0.z, s0.w, s1.x, s1.y, s1.z, s1.w,
                            s2.x, s2.y, s2.z, s2.w, s3.x, s3.y, s3.z, s3.w};
    v16h aH;
    #pragma unroll
    for (int i = 0; i < 16; ++i) aH[i] = (_Float16)ssp_f(th[i] + hsrc[i]);

    v8f n0 = {}, n1 = {};
    n0 = __builtin_amdgcn_wmma_f32_16x16x32_f16(false, aH, false, Bw2n0, (short)0, n0, false, false);
    n1 = __builtin_amdgcn_wmma_f32_16x16x32_f16(false, aH, false, Bw2n1, (short)0, n1, false, false);

    // ---- C) consume GEMM1: bias pass, ssp pass, store pass (batched chains) ----
    float xq[16];
    #pragma unroll
    for (int r = 0; r < 8; ++r) { xq[r] = c10[r] + b1e_c0; xq[r + 8] = c11[r] + b1e_c1; }
    _Float16 hq[16];
    #pragma unroll
    for (int i = 0; i < 16; ++i) hq[i] = (_Float16)ssp_f(xq[i]);
    #pragma unroll
    for (int r = 0; r < 8; ++r) {
      lds_g[wv][r + 8 * half][ln     ] = hq[r];
      lds_g[wv][r + 8 * half][ln + 16] = hq[r + 8];
    }
    // legalizer inserts s_wait_dscnt for the RAW through lds_g
    v8h glo = *(const v8h*)&lds_g[wv][ln][k0];
    v8h ghi = *(const v8h*)&lds_g[wv][ln][k0 + 16];
    v16h aG;
    #pragma unroll
    for (int i = 0; i < 8; ++i) { aG[i] = glo[i]; aG[i + 8] = ghi[i]; }

    // ---- GEMM2: ssp(...) x W2e -> raw gates ----
    v8f g0 = {}, g1 = {};
    g0 = __builtin_amdgcn_wmma_f32_16x16x32_f16(false, aG, false, Bw2e0, (short)0, g0, false, false);
    g1 = __builtin_amdgcn_wmma_f32_16x16x32_f16(false, aG, false, Bw2e1, (short)0, g1, false, false);

    // ---- D) fused mask * gate * nodes, reduced over this wave's a-slice ----
    const float4 w0 = *(const float4*)&lds_w[wv][a - a0][8 * half];      // rows r=0..3
    const float4 w1 = *(const float4*)&lds_w[wv][a - a0][8 * half + 4];  // rows r=4..7
    const float wr[8] = {w0.x, w0.y, w0.z, w0.w, w1.x, w1.y, w1.z, w1.w};
    #pragma unroll
    for (int r = 0; r < 8; ++r) {
      acc0[r] += wr[r] * (g0[r] + b2e_c0) * (n0[r] + b2n_c0);
      acc1[r] += wr[r] * (g1[r] + b2e_c1) * (n1[r] + b2n_c1);
    }
  }

  // ---- write deterministic per-slice partial (C-layout scatter) ----
  #pragma unroll
  for (int r = 0; r < 8; ++r) {
    float* __restrict__ o = part + (size_t)slice * OUT_ELEMS +
                            ((size_t)b * NP + p0 + r + 8 * half) * NF;
    o[ln     ] = acc0[r];
    o[ln + 16] = acc1[r];
  }
}

// fixed-order slice reduction -> bitwise deterministic output
__global__ void reduce_slices(const float* __restrict__ part, float* __restrict__ out) {
  const int idx = blockIdx.x * blockDim.x + threadIdx.x;
  if (idx < OUT_ELEMS) {
    float s = 0.f;
    #pragma unroll
    for (int i = 0; i < ASPLIT; ++i) s += part[(size_t)i * OUT_ELEMS + idx];
    out[idx] = s;
  }
}

extern "C" void kernel_launch(void* const* d_in, const int* in_sizes, int n_in,
                              void* d_out, int out_size, void* d_ws, size_t ws_size,
                              hipStream_t stream) {
  const float* scalar          = (const float*)d_in[0];
  const float* scalar_reciever = (const float*)d_in[1];
  const float* expansion       = (const float*)d_in[2];
  const float* mask            = (const float*)d_in[3];
  const float* edge_distance   = (const float*)d_in[4];
  const float* W1e = (const float*)d_in[5];
  const float* b1e = (const float*)d_in[6];
  const float* W2e = (const float*)d_in[7];
  const float* b2e = (const float*)d_in[8];
  const float* W1n = (const float*)d_in[9];
  const float* b1n = (const float*)d_in[10];
  const float* W2n = (const float*)d_in[11];
  const float* b2n = (const float*)d_in[12];
  float* out = (float*)d_out;

  float* tgt  = (float*)d_ws;                        // NB*NP*NF floats
  float* src  = tgt + (size_t)NB * NP * NF;          // NB*NA*NF floats
  float* part = src + (size_t)NB * NA * NF;          // ASPLIT * OUT_ELEMS floats

  const int pre_total = NB * NP * NF + NB * NA * NF;
  precompute_proj<<<(pre_total + 255) / 256, 256, 0, stream>>>(
      scalar, scalar_reciever, W1n, tgt, src);

  const int nwaves = NB * PT * ASPLIT;               // 2048
  deepdft_fused<<<nwaves / WAVES_PER_BLOCK, 32 * WAVES_PER_BLOCK, 0, stream>>>(
      expansion, mask, edge_distance, W1e, b1e, W2e, b2e, b1n, W2n, b2n, tgt, src, part);

  reduce_slices<<<(OUT_ELEMS + 255) / 256, 256, 0, stream>>>(part, out);
}
